// myGNN_39814346834540
// MI455X (gfx1250) — matrix operand order repaired and checked by reading, not compiled, not run on hardware
//
#include <hip/hip_runtime.h>
#include <hip/hip_bf16.h>
#include <math.h>

// Problem sizes (from reference): B=256, S=51, DIN=512, DEG=50
#define BB 256
#define SSEQ 51
#define NN (BB * SSEQ)      // 13056 nodes (== 204 * 64, == 408 * 32)
#define EE (NN * 50)        // 652800 edges (== 20400 * 32)

// Output layout (flat f32): A (N*128) | e (E) | pos (256*3) | ori (256*4)
#define A_OFF   0
#define E_OFF   (NN * 128)              // 1671168
#define POS_OFF (E_OFF + EE)            // 2323968
#define ORI_OFF (POS_OFF + BB * 3)      // 2324736

typedef __attribute__((ext_vector_type(16))) __bf16 v16bf;
typedef __attribute__((ext_vector_type(8)))  __bf16 v8bf;
typedef __attribute__((ext_vector_type(8)))  float  v8f;

// ---------- bf16 helpers (round-to-nearest-even via bit tricks) ----------
__device__ __forceinline__ __bf16 f2bf(float f) {
  unsigned u = __float_as_uint(f);
  unsigned r = (u + 0x7FFFu + ((u >> 16) & 1u)) >> 16;
  unsigned short h = (unsigned short)r;
  return __builtin_bit_cast(__bf16, h);
}
__device__ __forceinline__ float bf2f(__bf16 b) {
  unsigned short h = __builtin_bit_cast(unsigned short, b);
  return __uint_as_float(((unsigned)h) << 16);
}

// ---------- WMMA fragment loaders (v_wmma_f32_16x16x32_bf16) ----------
// A (16x32, row-major source): lane L: row = L%16; lanes 0-15 hold
// K = kb+0..7 (VGPR0-3) and kb+16..23 (VGPR4-7); lanes 16-31: +8.  (ISA 7.12.2)
__device__ __forceinline__ v16bf load_frag_a(const __bf16* base, int ld, int kb) {
  int lane = threadIdx.x & 31;
  int row  = lane & 15;
  int koff = (lane >> 4) << 3;            // 0 or 8
  const __bf16* p = base + row * ld + kb + koff;
  v8bf lo = *(const v8bf*)(p);
  v8bf hi = *(const v8bf*)(p + 16);
  v16bf f;
#pragma unroll
  for (int i = 0; i < 8; ++i) { f[i] = lo[i]; f[i + 8] = hi[i]; }
  return f;
}
// B (32x16), from pre-transposed weight Wt[n][k]: lane L: col = L%16;
// lanes 0-15 hold K = kb+0..7 and kb+16..23; lanes 16-31: +8.
__device__ __forceinline__ v16bf load_frag_b(const __bf16* base, int ld, int kb) {
  int lane = threadIdx.x & 31;
  int col  = lane & 15;
  int koff = (lane >> 4) << 3;            // 0 or 8
  const __bf16* p = base + col * ld + kb + koff;
  v8bf lo = *(const v8bf*)(p);
  v8bf hi = *(const v8bf*)(p + 16);
  v16bf f;
#pragma unroll
  for (int i = 0; i < 8; ++i) { f[i] = lo[i]; f[i + 8] = hi[i]; }
  return f;
}

// NT adjacent 16x16 output tiles: one shared A-fragment per K-step feeds NT
// independent accumulation chains (better XDL overlap, 1/NT A-frag traffic).
template <int NT>
__device__ __forceinline__ void gemm_tiles(const __bf16* A, int lda,
                                           const __bf16* Bt, int ldb,
                                           int K, v8f* acc) {
  for (int kb = 0; kb < K; kb += 32) {
    v16bf a = load_frag_a(A, lda, kb);
#pragma unroll
    for (int i = 0; i < NT; ++i) {
      v16bf b = load_frag_b(Bt + (size_t)(i * 16) * ldb, ldb, kb);
      acc[i] = __builtin_amdgcn_wmma_f32_16x16x32_bf16(false, a, false, b,
                                                       (short)0, acc[i],
                                                       false, false);
    }
  }
}

// C/D layout: lane L: col = L%16, rows = (L>=16 ? 8 : 0) + j for VGPR j.
// act: 0 = none, 1 = relu, 2 = leaky relu (0.01)
__device__ __forceinline__ void store_tile_bf(__bf16* out, int ldo,
                                              const float* bias, int colBase,
                                              v8f acc, int act) {
  int lane = threadIdx.x & 31;
  int col  = lane & 15;
  int rb   = (lane >> 4) << 3;
  float bv = bias[colBase + col];
#pragma unroll
  for (int j = 0; j < 8; ++j) {
    float v = acc[j] + bv;
    if (act == 1) v = fmaxf(v, 0.0f);
    else if (act == 2) v = (v > 0.0f) ? v : 0.01f * v;
    out[(rb + j) * ldo + col] = f2bf(v);
  }
}

// ---------- prep kernels ----------
__global__ void k_cvt_bf(const float* __restrict__ in, __bf16* __restrict__ out, int n) {
  int i = blockIdx.x * blockDim.x + threadIdx.x;
  if (i < n) out[i] = f2bf(in[i]);
}
// Wt[n*ldOut + kOff + k] = (bf16) W[k*Nc + n]
__global__ void k_transpose_bf(const float* __restrict__ W, __bf16* __restrict__ Wt,
                               int K, int Nc, int ldOut, int kOff) {
  int idx = blockIdx.x * blockDim.x + threadIdx.x;
  if (idx >= K * Nc) return;
  int k = idx / Nc, n = idx % Nc;
  Wt[(size_t)n * ldOut + kOff + k] = f2bf(W[idx]);
}

// ---------- fused node MLP: x(512) -> 256 -> 128 -> 64, relu, 64-row tiles ----
__global__ void __launch_bounds__(256) k_node(
    const __bf16* __restrict__ xbf,
    const __bf16* __restrict__ w1t, const float* __restrict__ b1,
    const __bf16* __restrict__ w2t, const float* __restrict__ b2,
    const __bf16* __restrict__ w3t, const float* __restrict__ b3,
    __bf16* __restrict__ hcat) {
  __shared__ __align__(16) __bf16 h1[64 * 272];   // 256 cols + 16 pad
  __shared__ __align__(16) __bf16 h2[64 * 144];   // 128 cols + 16 pad
  int rowBase = blockIdx.x * 64;
  int wave = threadIdx.x >> 5;
  // L1: 4 row-tiles x 16 col-tiles = 16 tasks of 4 tiles
  for (int t = wave; t < 16; t += 8) {
    int rt = t >> 2, cg = t & 3;
    v8f acc[4] = {};
    gemm_tiles<4>(xbf + (size_t)(rowBase + rt * 16) * 512, 512,
                  w1t + (size_t)(cg * 4 * 16) * 512, 512, 512, acc);
#pragma unroll
    for (int i = 0; i < 4; ++i) {
      int ct = cg * 4 + i;
      store_tile_bf(&h1[rt * 16 * 272 + ct * 16], 272, b1, ct * 16, acc[i], 1);
    }
  }
  __syncthreads();
  // L2: 4 x 8 = 8 tasks of 4 tiles
  {
    int rt = wave >> 1, cg = wave & 1;
    v8f acc[4] = {};
    gemm_tiles<4>(&h1[rt * 16 * 272], 272, w2t + (size_t)(cg * 4 * 16) * 256,
                  256, 256, acc);
#pragma unroll
    for (int i = 0; i < 4; ++i) {
      int ct = cg * 4 + i;
      store_tile_bf(&h2[rt * 16 * 144 + ct * 16], 144, b2, ct * 16, acc[i], 1);
    }
  }
  __syncthreads();
  // L3: 4 x 4 = 8 tasks of 2 tiles -> hcat cols 0..63
  {
    int rt = wave >> 1, cg = wave & 1;
    v8f acc[2] = {};
    gemm_tiles<2>(&h2[rt * 16 * 144], 144, w3t + (size_t)(cg * 2 * 16) * 128,
                  128, 128, acc);
#pragma unroll
    for (int i = 0; i < 2; ++i) {
      int ct = cg * 2 + i;
      store_tile_bf(hcat + (size_t)(rowBase + rt * 16) * 128 + ct * 16, 128,
                    b3, ct * 16, acc[i], 1);
    }
  }
}

// ---------- pose MLP: 7 -> 16 -> 32 -> 64 (tiny; VALU) -> hcat cols 64..127 --
__global__ void k_pose(const float* __restrict__ xp,
                       const float* __restrict__ w1, const float* __restrict__ b1,
                       const float* __restrict__ w2, const float* __restrict__ b2,
                       const float* __restrict__ w3, const float* __restrict__ b3,
                       __bf16* __restrict__ hcat) {
  int n = blockIdx.x * blockDim.x + threadIdx.x;
  if (n >= NN) return;
  float in[7];
#pragma unroll
  for (int j = 0; j < 7; ++j) in[j] = xp[n * 7 + j];
  float t1[16];
  for (int o = 0; o < 16; ++o) {
    float a = b1[o];
#pragma unroll
    for (int j = 0; j < 7; ++j) a += in[j] * w1[j * 16 + o];
    t1[o] = fmaxf(a, 0.0f);
  }
  float t2[32];
  for (int o = 0; o < 32; ++o) {
    float a = b2[o];
#pragma unroll
    for (int j = 0; j < 16; ++j) a += t1[j] * w2[j * 32 + o];
    t2[o] = fmaxf(a, 0.0f);
  }
  for (int o = 0; o < 64; ++o) {
    float a = b3[o];
#pragma unroll
    for (int j = 0; j < 32; ++j) a += t2[j] * w3[j * 64 + o];
    hcat[(size_t)n * 128 + 64 + o] = f2bf(fmaxf(a, 0.0f));
  }
}

// ---------- encoder: hcat(128) -> 256, relu; writes xn f32 + ncat bf16 -------
__global__ void __launch_bounds__(256) k_enc(
    const __bf16* __restrict__ hcat, const __bf16* __restrict__ wt,
    const float* __restrict__ bias, float* __restrict__ xn,
    __bf16* __restrict__ ncat) {
  int rowBase = blockIdx.x * 64;
  int wave = threadIdx.x >> 5;
  // 4 row-tiles x 16 col-tiles = 16 tasks of 4 tiles
  for (int t = wave; t < 16; t += 8) {
    int rt = t >> 2, cg = t & 3;
    v8f acc[4] = {};
    gemm_tiles<4>(hcat + (size_t)(rowBase + rt * 16) * 128, 128,
                  wt + (size_t)(cg * 4 * 16) * 128, 128, 128, acc);
    int lane = threadIdx.x & 31, col = lane & 15, rb = (lane >> 4) << 3;
#pragma unroll
    for (int i = 0; i < 4; ++i) {
      int ct = cg * 4 + i;
      float bv = bias[ct * 16 + col];
#pragma unroll
      for (int j = 0; j < 8; ++j) {
        float v = fmaxf(acc[i][j] + bv, 0.0f);
        int gr = rowBase + rt * 16 + rb + j;
        xn[(size_t)gr * 256 + ct * 16 + col] = v;
        ncat[(size_t)gr * 512 + ct * 16 + col] = f2bf(v);  // left half of concat
      }
    }
  }
}

// ---------- fused edge MLP: d=xn[s]-xn[d]; 256->256->128->64->1; msg atomics -
__global__ void __launch_bounds__(256) k_edge(
    const float* __restrict__ xn, const int* __restrict__ esrc,
    const int* __restrict__ edst,
    const __bf16* __restrict__ ew1t, const float* __restrict__ eb1,
    const __bf16* __restrict__ ew2t, const float* __restrict__ eb2,
    const __bf16* __restrict__ ew3t, const float* __restrict__ eb3,
    const float* __restrict__ ew4, const float* __restrict__ eb4,
    float* __restrict__ eout, float* __restrict__ nsum, float* __restrict__ deg) {
  __shared__ __align__(16) __bf16 dl[32 * 272];
  __shared__ __align__(16) __bf16 s1[32 * 272];
  __shared__ __align__(16) __bf16 s2[32 * 144];
  __shared__ __align__(16) __bf16 s3[32 * 80];
  __shared__ float  ep[32];
  __shared__ int    isrc[32], idst[32];
  int tid  = threadIdx.x;
  int base = blockIdx.x * 32;
  if (tid < 32) { isrc[tid] = esrc[base + tid]; idst[tid] = edst[base + tid]; }
  __syncthreads();
  {  // stage d tile: 8 threads per edge, 32 cols each
    int r = tid >> 3, cp = (tid & 7) << 5;
    const float* ps = xn + (size_t)isrc[r] * 256 + cp;
    const float* pd = xn + (size_t)idst[r] * 256 + cp;
#pragma unroll 8
    for (int c = 0; c < 32; ++c) dl[r * 272 + cp + c] = f2bf(ps[c] - pd[c]);
  }
  __syncthreads();
  int wave = tid >> 5;
  // L1: 2 row-tiles x 16 col-tiles = 8 tasks of 4 tiles
  {
    int rt = wave >> 2, cg = wave & 3;
    v8f acc[4] = {};
    gemm_tiles<4>(&dl[rt * 16 * 272], 272, ew1t + (size_t)(cg * 4 * 16) * 256,
                  256, 256, acc);
#pragma unroll
    for (int i = 0; i < 4; ++i) {
      int ct = cg * 4 + i;
      store_tile_bf(&s1[rt * 16 * 272 + ct * 16], 272, eb1, ct * 16, acc[i], 2);
    }
  }
  __syncthreads();
  // L2: 2 x 8 = 8 tasks of 2 tiles
  {
    int rt = wave >> 2, cg = wave & 3;
    v8f acc[2] = {};
    gemm_tiles<2>(&s1[rt * 16 * 272], 272, ew2t + (size_t)(cg * 2 * 16) * 256,
                  256, 256, acc);
#pragma unroll
    for (int i = 0; i < 2; ++i) {
      int ct = cg * 2 + i;
      store_tile_bf(&s2[rt * 16 * 144 + ct * 16], 144, eb2, ct * 16, acc[i], 2);
    }
  }
  __syncthreads();
  // L3: 2 x 4 = 8 tasks of 1 tile
  {
    int rt = wave >> 2, ct = wave & 3;
    v8f acc[1] = {};
    gemm_tiles<1>(&s2[rt * 16 * 144], 144, ew3t + (size_t)(ct * 16) * 128,
                  128, 128, acc);
    store_tile_bf(&s3[rt * 16 * 80 + ct * 16], 80, eb3, ct * 16, acc[0], 2);
  }
  __syncthreads();
  if (tid < 32) {                               // L4 + sigmoid + degree
    float a = eb4[0];
#pragma unroll
    for (int k = 0; k < 64; ++k) a += bf2f(s3[tid * 80 + k]) * ew4[k];
    float ev = 1.0f / (1.0f + expf(-a));
    ep[tid] = ev;
    eout[base + tid] = ev;
    atomicAdd(deg + idst[tid], 1.0f);
  }
  __syncthreads();
  {  // message scatter: msg = e * xn[src] into neigh_sum[dst]
    int r = tid >> 3, cp = (tid & 7) << 5;
    float ev = ep[r];
    const float* ps = xn + (size_t)isrc[r] * 256 + cp;
    float* pn = nsum + (size_t)idst[r] * 256 + cp;
    for (int c = 0; c < 32; ++c) atomicAdd(pn + c, ev * ps[c]);
  }
}

// ---------- neigh = neigh_sum / max(deg,1) -> right half of ncat -------------
__global__ void k_neigh(const float* __restrict__ ns, const float* __restrict__ deg,
                        __bf16* __restrict__ ncat) {
  int idx = blockIdx.x * blockDim.x + threadIdx.x;
  if (idx >= NN * 256) return;
  int nrow = idx >> 8, c = idx & 255;
  float dg = fmaxf(deg[nrow], 1.0f);
  ncat[(size_t)nrow * 512 + 256 + c] = f2bf(ns[idx] / dg);
}

// ---------- SAGE: A = [xn||neigh](512) @ [wself;wneigh](512x128) + b ---------
// A-operand tile staged into LDS with CDNA5 async global->LDS copies
// (GLOBAL_LOAD_ASYNC_TO_LDS_B128, ASYNCcnt), padded ld to dodge bank conflicts.
__global__ void __launch_bounds__(256) k_sage(
    const __bf16* __restrict__ ncat, const __bf16* __restrict__ wt,
    const float* __restrict__ sageb, float* __restrict__ Araw) {
  __shared__ __align__(16) __bf16 at[32 * 520];   // 32 rows, 512 cols + 8 pad
  int rowBase = blockIdx.x * 32;
  // LDS byte address = low 32 bits of the generic shared-aperture address
  unsigned lds_base = (unsigned)(unsigned long long)(const void*)at;
  for (int i = threadIdx.x; i < 32 * 64; i += 256) {   // 16B chunks
    int r = i >> 6, c = i & 63;
    unsigned lds_off = lds_base + (unsigned)(r * 1040 + c * 16);
    unsigned long long ga =
        (unsigned long long)(const void*)(ncat + (size_t)(rowBase + r) * 512 + c * 8);
    asm volatile("global_load_async_to_lds_b128 %0, %1, off"
                 :: "v"(lds_off), "v"(ga) : "memory");
  }
  asm volatile("s_wait_asynccnt 0x0" ::: "memory");
  __syncthreads();
  int wave = threadIdx.x >> 5;
  // 2 row-tiles x 8 col-tiles = 8 tasks of 2 tiles
  int rt = wave >> 2, cg = wave & 3;
  v8f acc[2] = {};
  gemm_tiles<2>(&at[rt * 16 * 520], 520, wt + (size_t)(cg * 2 * 16) * 512,
                512, 512, acc);
  int lane = threadIdx.x & 31, col = lane & 15, rb = (lane >> 4) << 3;
#pragma unroll
  for (int i = 0; i < 2; ++i) {
    int ct = cg * 2 + i;
    float bv = sageb[ct * 16 + col];
#pragma unroll
    for (int j = 0; j < 8; ++j)
      Araw[(size_t)(rowBase + rt * 16 + rb + j) * 128 + ct * 16 + col] =
          acc[i][j] + bv;
  }
}

// ---------- decoder: est_pose only needed at s==0 (256 rows) -----------------
__global__ void k_dec(const float* __restrict__ Araw, const float* __restrict__ decw,
                      const float* __restrict__ decb, float* __restrict__ out) {
  int b = blockIdx.x * blockDim.x + threadIdx.x;
  if (b >= BB) return;
  const float* a = Araw + (size_t)(b * SSEQ) * 128;
  float est[7];
#pragma unroll
  for (int j = 0; j < 7; ++j) est[j] = decb[j];
  for (int k = 0; k < 128; ++k) {
    float av = a[k];
#pragma unroll
    for (int j = 0; j < 7; ++j) est[j] += av * decw[k * 7 + j];
  }
  out[POS_OFF + b * 3 + 0] = est[0];
  out[POS_OFF + b * 3 + 1] = est[1];
  out[POS_OFF + b * 3 + 2] = est[2];
  out[ORI_OFF + b * 4 + 0] = est[3];
  out[ORI_OFF + b * 4 + 1] = est[4];
  out[ORI_OFF + b * 4 + 2] = est[5];
  out[ORI_OFF + b * 4 + 3] = est[6];
}

// ---------- A = lrelu(A); A /= max(||A||_(axis=S), eps) ----------------------
__global__ void k_final(const float* __restrict__ Araw, float* __restrict__ out) {
  int b = blockIdx.x;         // batch
  int c = threadIdx.x;        // channel 0..127
  float ss = 0.0f;
  for (int s = 0; s < SSEQ; ++s) {
    float v = Araw[((size_t)(b * SSEQ + s)) * 128 + c];
    v = (v > 0.0f) ? v : 0.01f * v;
    ss += v * v;
  }
  float inv = 1.0f / fmaxf(sqrtf(ss), 1e-12f);
  for (int s = 0; s < SSEQ; ++s) {
    float v = Araw[((size_t)(b * SSEQ + s)) * 128 + c];
    v = (v > 0.0f) ? v : 0.01f * v;
    out[A_OFF + ((size_t)(b * SSEQ + s)) * 128 + c] = v * inv;
  }
}

extern "C" void kernel_launch(void* const* d_in, const int* in_sizes, int n_in,
                              void* d_out, int out_size, void* d_ws, size_t ws_size,
                              hipStream_t stream) {
  (void)in_sizes; (void)n_in; (void)out_size; (void)ws_size;
  const float* x      = (const float*)d_in[0];
  const float* xpose  = (const float*)d_in[1];
  const float* m2w1   = (const float*)d_in[2];
  const float* m2b1   = (const float*)d_in[3];
  const float* m2w2   = (const float*)d_in[4];
  const float* m2b2   = (const float*)d_in[5];
  const float* m2w3   = (const float*)d_in[6];
  const float* m2b3   = (const float*)d_in[7];
  const float* m3w1   = (const float*)d_in[8];
  const float* m3b1   = (const float*)d_in[9];
  const float* m3w2   = (const float*)d_in[10];
  const float* m3b2   = (const float*)d_in[11];
  const float* m3w3   = (const float*)d_in[12];
  const float* m3b3   = (const float*)d_in[13];
  const float* encw   = (const float*)d_in[14];
  const float* encb   = (const float*)d_in[15];
  const float* ew1    = (const float*)d_in[16];
  const float* eb1    = (const float*)d_in[17];
  const float* ew2    = (const float*)d_in[18];
  const float* eb2    = (const float*)d_in[19];
  const float* ew3    = (const float*)d_in[20];
  const float* eb3    = (const float*)d_in[21];
  const float* ew4    = (const float*)d_in[22];
  const float* eb4    = (const float*)d_in[23];
  const float* wself  = (const float*)d_in[24];
  const float* wneigh = (const float*)d_in[25];
  const float* sageb  = (const float*)d_in[26];
  const float* decw   = (const float*)d_in[27];
  const float* decb   = (const float*)d_in[28];
  const int*   esrc   = (const int*)d_in[29];
  const int*   edst   = (const int*)d_in[30];
  float* out = (float*)d_out;

  char* ws = (char*)d_ws;
  size_t off = 0;
  auto carve = [&](size_t bytes) -> void* {
    void* p = (void*)(ws + off);
    off += (bytes + 255) & ~(size_t)255;
    return p;
  };
  __bf16* xbf   = (__bf16*)carve((size_t)NN * 512 * 2);
  __bf16* hcat  = (__bf16*)carve((size_t)NN * 128 * 2);
  float*  xn    = (float*) carve((size_t)NN * 256 * 4);
  __bf16* ncat  = (__bf16*)carve((size_t)NN * 512 * 2);
  float*  nsum  = (float*) carve((size_t)NN * 256 * 4);
  float*  deg   = (float*) carve((size_t)NN * 4);
  float*  Araw  = (float*) carve((size_t)NN * 128 * 4);
  __bf16* w1t   = (__bf16*)carve((size_t)256 * 512 * 2);
  __bf16* w2t   = (__bf16*)carve((size_t)128 * 256 * 2);
  __bf16* w3t   = (__bf16*)carve((size_t)64 * 128 * 2);
  __bf16* encwt = (__bf16*)carve((size_t)256 * 128 * 2);
  __bf16* ew1t  = (__bf16*)carve((size_t)256 * 256 * 2);
  __bf16* ew2t  = (__bf16*)carve((size_t)128 * 256 * 2);
  __bf16* ew3t  = (__bf16*)carve((size_t)64 * 128 * 2);
  __bf16* sgwt  = (__bf16*)carve((size_t)128 * 512 * 2);

  // --- prep: bf16 input copy + transposed bf16 weights ---
  k_cvt_bf<<<(NN * 512 + 255) / 256, 256, 0, stream>>>(x, xbf, NN * 512);
  k_transpose_bf<<<(512 * 256 + 255) / 256, 256, 0, stream>>>(m2w1, w1t, 512, 256, 512, 0);
  k_transpose_bf<<<(256 * 128 + 255) / 256, 256, 0, stream>>>(m2w2, w2t, 256, 128, 256, 0);
  k_transpose_bf<<<(128 * 64 + 255) / 256, 256, 0, stream>>>(m2w3, w3t, 128, 64, 128, 0);
  k_transpose_bf<<<(128 * 256 + 255) / 256, 256, 0, stream>>>(encw, encwt, 128, 256, 128, 0);
  k_transpose_bf<<<(256 * 256 + 255) / 256, 256, 0, stream>>>(ew1, ew1t, 256, 256, 256, 0);
  k_transpose_bf<<<(256 * 128 + 255) / 256, 256, 0, stream>>>(ew2, ew2t, 256, 128, 256, 0);
  k_transpose_bf<<<(128 * 64 + 255) / 256, 256, 0, stream>>>(ew3, ew3t, 128, 64, 128, 0);
  k_transpose_bf<<<(256 * 128 + 255) / 256, 256, 0, stream>>>(wself, sgwt, 256, 128, 512, 0);
  k_transpose_bf<<<(256 * 128 + 255) / 256, 256, 0, stream>>>(wneigh, sgwt, 256, 128, 512, 256);

  // --- node pipeline ---
  k_node<<<NN / 64, 256, 0, stream>>>(xbf, w1t, m2b1, w2t, m2b2, w3t, m2b3, hcat);
  k_pose<<<(NN + 255) / 256, 256, 0, stream>>>(xpose, m3w1, m3b1, m3w2, m3b2, m3w3, m3b3, hcat);
  k_enc<<<NN / 64, 256, 0, stream>>>(hcat, encwt, encb, xn, ncat);

  // --- edge pipeline (accumulators zeroed each call; capturable memsets) ---
  hipMemsetAsync(nsum, 0, (size_t)NN * 256 * 4, stream);
  hipMemsetAsync(deg, 0, (size_t)NN * 4, stream);
  k_edge<<<EE / 32, 256, 0, stream>>>(xn, esrc, edst, ew1t, eb1, ew2t, eb2,
                                      ew3t, eb3, ew4, eb4,
                                      out + E_OFF, nsum, deg);

  // --- aggregation + heads ---
  k_neigh<<<(NN * 256 + 255) / 256, 256, 0, stream>>>(nsum, deg, ncat);
  k_sage<<<NN / 32, 256, 0, stream>>>(ncat, sgwt, sageb, Araw);
  k_dec<<<1, 256, 0, stream>>>(Araw, decw, decb, out);
  k_final<<<BB, 128, 0, stream>>>(Araw, out);
}
